// ViGBlock_8383776161886
// MI455X (gfx1250) — matrix operand order repaired and checked by reading, not compiled, not run on hardware
//
#include <hip/hip_runtime.h>
#include <hip/hip_bf16.h>

typedef __attribute__((ext_vector_type(16))) __bf16 v16bf;
typedef __attribute__((ext_vector_type(8)))  __bf16 v8bf;
typedef __attribute__((ext_vector_type(8)))  float  v8f;

#define LRELU(v) ((v) > 0.0f ? (v) : 0.01f * (v))
#define NEG_INF_F (-1e30f)

// ---------------------------------------------------------------------------
// bf16 WMMA GEMM, operands direct from global (L2-resident weights, bf16
// shadow activations). out[M,N] = A[M,K] @ WT[N,K]^T + bias (+res / +leaky).
// Block: 256 threads = 8 waves (2x4). Block tile 32(M) x 256(N);
// each wave: one 16x16 A-row-tile x 4 N-tiles -> 4 accumulators.
// MODE 0: +bias    MODE 1: +bias+res    MODE 2: leaky(+bias)
// ---------------------------------------------------------------------------
template <int MODE>
__global__ __launch_bounds__(256) void gemm_bf16_kernel(
    const __bf16* __restrict__ A,    // [M, Kd] bf16
    const __bf16* __restrict__ WT,   // [Nd, Kd] bf16 (pre-transposed weights)
    const float* __restrict__ bias,
    const float* __restrict__ res,
    float* __restrict__ out,
    __bf16* __restrict__ outbf,      // optional bf16 shadow of out
    int Kd, int Nd) {
  const int tid = threadIdx.x;
  const int lane = tid & 31, wave = tid >> 5;
  const int wm = wave >> 2, wn = wave & 3;   // 2 x 4 wave grid
  const int lhalf = lane >> 4, lmod = lane & 15;
  const int r0 = blockIdx.y * 32;
  const int c0 = blockIdx.x * 256;

  const __bf16* Arow = A + (size_t)(r0 + wm * 16 + lmod) * Kd;
  const __bf16* Brow0 = WT + (size_t)(c0 + wn * 16 + lmod) * Kd;

  v8f acc0 = {}, acc1 = {}, acc2 = {}, acc3 = {};
  for (int kk = 0; kk < Kd; kk += 32) {
    // A fragment (16x32 bf16): lane-half selects K+16; contiguous 16 along K.
    v8bf alo = *(const v8bf*)(Arow + kk + (lhalf << 4));
    v8bf ahi = *(const v8bf*)(Arow + kk + 8 + (lhalf << 4));
    v16bf af = __builtin_shufflevector(alo, ahi, 0, 1, 2, 3, 4, 5, 6, 7,
                                       8, 9, 10, 11, 12, 13, 14, 15);
    // B fragments (32x16 bf16): lane-half selects K+16; contiguous 16 along K.
    v16bf b0 = *(const v16bf*)(Brow0 + (size_t)0 * 64 * Kd + kk + (lhalf << 4));
    v16bf b1 = *(const v16bf*)(Brow0 + (size_t)1 * 64 * Kd + kk + (lhalf << 4));
    v16bf b2 = *(const v16bf*)(Brow0 + (size_t)2 * 64 * Kd + kk + (lhalf << 4));
    v16bf b3 = *(const v16bf*)(Brow0 + (size_t)3 * 64 * Kd + kk + (lhalf << 4));
    acc0 = __builtin_amdgcn_wmma_f32_16x16x32_bf16(false, af, false, b0, (short)0, acc0, false, false);
    acc1 = __builtin_amdgcn_wmma_f32_16x16x32_bf16(false, af, false, b1, (short)0, acc1, false, false);
    acc2 = __builtin_amdgcn_wmma_f32_16x16x32_bf16(false, af, false, b2, (short)0, acc2, false, false);
    acc3 = __builtin_amdgcn_wmma_f32_16x16x32_bf16(false, af, false, b3, (short)0, acc3, false, false);
  }

  v8f accs[4] = {acc0, acc1, acc2, acc3};
  #pragma unroll
  for (int j = 0; j < 4; ++j) {
    const int col = c0 + j * 64 + wn * 16 + lmod;
    const float bv = bias[col];
    #pragma unroll
    for (int r = 0; r < 8; ++r) {
      int row = r0 + wm * 16 + (lhalf << 3) + r;  // C/D: m = r + 8*lane_half
      float v = accs[j][r] + bv;
      if (MODE == 1) v += res[(size_t)row * Nd + col];
      if (MODE == 2) v = LRELU(v);
      out[(size_t)row * Nd + col] = v;
      if (outbf) outbf[(size_t)row * Nd + col] = (__bf16)v;
    }
  }
}

// ---------------------------------------------------------------------------
// Row LayerNorm (+ gamma/beta) + LeakyReLU, in place; emits bf16 shadow.
// ---------------------------------------------------------------------------
__global__ __launch_bounds__(256) void ln_leaky_kernel(
    float* __restrict__ H, const float* __restrict__ g,
    const float* __restrict__ beta, __bf16* __restrict__ Hbf, int hidden) {
  __shared__ float red[256];
  float* row = H + (size_t)blockIdx.x * hidden;
  __bf16* rowbf = Hbf + (size_t)blockIdx.x * hidden;
  float sum = 0.f, sumsq = 0.f;
  for (int c = threadIdx.x; c < hidden; c += 256) {
    float v = row[c];
    sum += v; sumsq += v * v;
  }
  red[threadIdx.x] = sum; __syncthreads();
  for (int s = 128; s > 0; s >>= 1) {
    if (threadIdx.x < s) red[threadIdx.x] += red[threadIdx.x + s];
    __syncthreads();
  }
  float mu = red[0] / hidden;
  __syncthreads();
  red[threadIdx.x] = sumsq; __syncthreads();
  for (int s = 128; s > 0; s >>= 1) {
    if (threadIdx.x < s) red[threadIdx.x] += red[threadIdx.x + s];
    __syncthreads();
  }
  float var = red[0] / hidden - mu * mu;
  float inv = rsqrtf(var + 1e-5f);
  for (int c = threadIdx.x; c < hidden; c += 256) {
    float v = (row[c] - mu) * inv * g[c] + beta[c];
    v = LRELU(v);
    row[c] = v;
    rowbf[c] = (__bf16)v;
  }
}

// ---------------------------------------------------------------------------
// Fused similarity (WMMA) + top-K + max-aggregation + interleaved feature.
// One block per (batch, 16-row n-tile). 256 threads = 8 waves.
// A fragments (8 K-steps) held in registers; 2 m-tiles in flight per wave.
// ---------------------------------------------------------------------------
__global__ __launch_bounds__(256) void sim_topk_agg_kernel(
    const __bf16* __restrict__ x1bf, const float* __restrict__ x1,
    const unsigned char* __restrict__ mask, float* __restrict__ yin,
    __bf16* __restrict__ yinbf) {
  const int N = 1024, C = 256;
  const int b = blockIdx.y;
  const int n0 = blockIdx.x * 16;
  const int tid = threadIdx.x;
  const int lane = tid & 31, wave = tid >> 5;
  const int lhalf = lane >> 4, lmod = lane & 15;

  __shared__ float S[16][520];          // 16 n-rows x 512 m-cols (half of N)
  __shared__ float cval[16][16][9];
  __shared__ int   cidx[16][16][9];
  __shared__ int   graph[16][9];
  __shared__ unsigned char mall[1024];  // this batch's mask row

  const __bf16* xb = x1bf + (size_t)b * N * C;

  // stage mask row in LDS
  for (int i = tid; i < N; i += 256) mall[i] = mask[(size_t)b * N + i];

  // Preload all 8 A fragments (rows n0..n0+15, K=256) into registers.
  const __bf16* Aq = xb + (size_t)(n0 + lmod) * C;
  v16bf afs[8];
  #pragma unroll
  for (int s = 0; s < 8; ++s) {
    v8bf alo = *(const v8bf*)(Aq + s * 32 + (lhalf << 4));
    v8bf ahi = *(const v8bf*)(Aq + s * 32 + 8 + (lhalf << 4));
    afs[s] = __builtin_shufflevector(alo, ahi, 0, 1, 2, 3, 4, 5, 6, 7,
                                     8, 9, 10, 11, 12, 13, 14, 15);
  }
  __syncthreads();

  float tv[9]; int ti[9];
  #pragma unroll
  for (int k = 0; k < 9; ++k) { tv[k] = -3e38f; ti[k] = 0; }
  const int trow = tid >> 4, tslot = tid & 15;

  for (int half = 0; half < 2; ++half) {
    const int mbase = half * 512;
    // 32 m-tiles per half; each wave: 2 pair-iterations x 2 tiles in flight.
    for (int t = 0; t < 2; ++t) {
      const int m0a = mbase + ((2 * t) * 8 + wave) * 16;
      const int m0b = mbase + ((2 * t + 1) * 8 + wave) * 16;
      const __bf16* Brow0 = xb + (size_t)(m0a + lmod) * C;
      const __bf16* Brow1 = xb + (size_t)(m0b + lmod) * C;
      v8f acc0 = {}, acc1 = {};
      #pragma unroll
      for (int s = 0; s < 8; ++s) {
        v16bf b0 = *(const v16bf*)(Brow0 + s * 32 + (lhalf << 4));
        v16bf b1 = *(const v16bf*)(Brow1 + s * 32 + (lhalf << 4));
        acc0 = __builtin_amdgcn_wmma_f32_16x16x32_bf16(
            false, afs[s], false, b0, (short)0, acc0, false, false);
        acc1 = __builtin_amdgcn_wmma_f32_16x16x32_bf16(
            false, afs[s], false, b1, (short)0, acc1, false, false);
      }
      // Post-process + scatter into LDS score buffer (both tiles).
      const int ms[2] = {m0a + lmod, m0b + lmod};
      const v8f as[2] = {acc0, acc1};
      #pragma unroll
      for (int q = 0; q < 2; ++q) {
        const int m = ms[q];
        const bool mm = mall[m] != 0;
        #pragma unroll
        for (int r = 0; r < 8; ++r) {
          int nl = r + (lhalf << 3);
          int n = n0 + nl;
          bool mn = mall[n] != 0;
          float s = as[q][r];
          if (!(mn && mm)) s = NEG_INF_F;
          if (!mn && n == m) s = 0.0f;
          S[nl][m - mbase] = s;
        }
      }
    }
    __syncthreads();
    // Partial top-9 scan (32 columns per thread per half).
    for (int i = 0; i < 32; ++i) {
      int ml = tslot + (i << 4);
      float s = S[trow][ml];
      int gm = mbase + ml;
      if (s > tv[8]) {
        int p = 8;
        while (p > 0 && s > tv[p - 1]) { tv[p] = tv[p - 1]; ti[p] = ti[p - 1]; --p; }
        tv[p] = s; ti[p] = gm;
      }
    }
    __syncthreads();
  }

  #pragma unroll
  for (int k = 0; k < 9; ++k) { cval[trow][tslot][k] = tv[k]; cidx[trow][tslot][k] = ti[k]; }
  __syncthreads();

  // Merge 16x9 candidates per row -> final top-9 (ties: lowest index).
  if (tid < 16) {
    int row = tid;
    for (int k = 0; k < 9; ++k) {
      float best = -3e38f; int bi = 0x7fffffff, bs = 0, bk = 0;
      for (int s2 = 0; s2 < 16; ++s2)
        for (int k2 = 0; k2 < 9; ++k2) {
          float v = cval[row][s2][k2]; int id = cidx[row][s2][k2];
          if (v > best || (v == best && id < bi)) { best = v; bi = id; bs = s2; bk = k2; }
        }
      graph[row][k] = bi;
      cval[row][bs][bk] = -3e38f;
    }
  }
  __syncthreads();

  // agg = max_k(x1[g_k]) - x1[n]; interleaved: y[2c]=x1, y[2c+1]=agg.
  {
    const int n = n0 + trow;
    const float* xrow = x1 + ((size_t)b * N + n) * C;
    float* yrow = yin + ((size_t)b * N + n) * (2 * C);
    __bf16* ybrow = yinbf + ((size_t)b * N + n) * (2 * C);
    int g[9];
    #pragma unroll
    for (int k = 0; k < 9; ++k) g[k] = graph[trow][k];
    for (int c = tslot * 16; c < tslot * 16 + 16; ++c) {
      float base = xrow[c];
      float mx = -3e38f;
      #pragma unroll
      for (int k = 0; k < 9; ++k)
        mx = fmaxf(mx, x1[((size_t)b * N + g[k]) * C + c]);
      float ag = mx - base;
      yrow[2 * c] = base;
      yrow[2 * c + 1] = ag;
      ybrow[2 * c] = (__bf16)base;
      ybrow[2 * c + 1] = (__bf16)ag;
    }
  }
}

// ---------------------------------------------------------------------------
// Elementwise / prep helpers
// ---------------------------------------------------------------------------
__global__ void wtrans_kernel(const float* __restrict__ W, __bf16* __restrict__ WT,
                              int Kd, int Nd) {
  size_t i = (size_t)blockIdx.x * blockDim.x + threadIdx.x;
  if (i >= (size_t)Kd * Nd) return;
  int n = (int)(i / Kd), k = (int)(i % Kd);
  WT[i] = (__bf16)W[(size_t)k * Nd + n];
}

__global__ void f2bf_kernel(const float* __restrict__ a, __bf16* __restrict__ o) {
  size_t i = (size_t)blockIdx.x * blockDim.x + threadIdx.x;
  o[i] = (__bf16)a[i];
}

__global__ void mask_convert_kernel(float* __restrict__ x1,
                                    const unsigned char* __restrict__ mask,
                                    __bf16* __restrict__ x1bf, int Cc) {
  size_t i = (size_t)blockIdx.x * blockDim.x + threadIdx.x;
  size_t row = i / Cc;
  float v = mask[row] ? x1[i] : 0.0f;
  x1[i] = v;
  x1bf[i] = (__bf16)v;
}

__global__ void add_kernel(const float* __restrict__ a, const float* __restrict__ b,
                           float* __restrict__ o, __bf16* __restrict__ obf) {
  size_t i = (size_t)blockIdx.x * blockDim.x + threadIdx.x;
  float v = a[i] + b[i];
  o[i] = v;
  obf[i] = (__bf16)v;
}

__global__ void leaky_kernel(const float* __restrict__ a, float* __restrict__ o,
                             __bf16* __restrict__ obf) {
  size_t i = (size_t)blockIdx.x * blockDim.x + threadIdx.x;
  float v = a[i];
  v = LRELU(v);
  o[i] = v;
  obf[i] = (__bf16)v;
}

__global__ void final_kernel(const float* __restrict__ h2,
                             const float* __restrict__ x2,
                             const unsigned char* __restrict__ mask,
                             float* __restrict__ o, int Cc) {
  size_t i = (size_t)blockIdx.x * blockDim.x + threadIdx.x;
  size_t row = i / Cc;
  o[i] = mask[row] ? (h2[i] + x2[i]) : 0.0f;
}

// ---------------------------------------------------------------------------
// Host orchestration
// ---------------------------------------------------------------------------
extern "C" void kernel_launch(void* const* d_in, const int* in_sizes, int n_in,
                              void* d_out, int out_size, void* d_ws, size_t ws_size,
                              hipStream_t stream) {
  (void)in_sizes; (void)n_in; (void)out_size; (void)ws_size;
  const int B = 32, N = 1024, Cc = 256, M = B * N;

  const float* x = (const float*)d_in[0];
  const unsigned char* mask = (const unsigned char*)d_in[1];
  const float* fc_W = (const float*)d_in[2];
  const float* fc_b = (const float*)d_in[3];
  const float* il1[6]; const float* ol1[6]; const float* il2[6]; const float* ol2[6];
  for (int i = 0; i < 6; ++i) {
    il1[i] = (const float*)d_in[4 + i];
    ol1[i] = (const float*)d_in[10 + i];
    il2[i] = (const float*)d_in[16 + i];
    ol2[i] = (const float*)d_in[22 + i];
  }
  float* outp = (float*)d_out;

  // ---- workspace layout ----
  char* ws = (char*)d_ws;
  size_t off = 0;
  auto alloc = [&](size_t bytes) -> void* {
    void* p = ws + off;
    off += (bytes + 255) & ~(size_t)255;
    return p;
  };
  float*  x1  = (float*)alloc((size_t)M * Cc * 4);
  float*  Hb  = (float*)alloc((size_t)M * 1024 * 4);
  float*  yin = (float*)alloc((size_t)M * 512 * 4);
  float*  yl  = (float*)alloc((size_t)M * Cc * 4);
  float*  tb  = (float*)alloc((size_t)M * Cc * 4);
  float*  x2  = (float*)alloc((size_t)M * Cc * 4);
  float*  hb  = (float*)alloc((size_t)M * Cc * 4);
  float*  hl  = (float*)alloc((size_t)M * Cc * 4);
  float*  h2  = (float*)alloc((size_t)M * Cc * 4);
  __bf16* xbf   = (__bf16*)alloc((size_t)M * Cc * 2);
  __bf16* x1bf  = (__bf16*)alloc((size_t)M * Cc * 2);
  __bf16* Hbf   = (__bf16*)alloc((size_t)M * 1024 * 2);
  __bf16* yinbf = (__bf16*)alloc((size_t)M * 512 * 2);
  __bf16* ylbf  = (__bf16*)alloc((size_t)M * Cc * 2);
  __bf16* x2bf  = (__bf16*)alloc((size_t)M * Cc * 2);
  __bf16* hlbf  = (__bf16*)alloc((size_t)M * Cc * 2);
  __bf16* il1W1T = (__bf16*)alloc((size_t)Cc * Cc * 2);
  __bf16* il1W2T = (__bf16*)alloc((size_t)Cc * Cc * 2);
  __bf16* ol1W1T = (__bf16*)alloc((size_t)Cc * Cc * 2);
  __bf16* ol1W2T = (__bf16*)alloc((size_t)Cc * Cc * 2);
  __bf16* fcWT   = (__bf16*)alloc((size_t)Cc * 2 * Cc * 2);
  __bf16* il2W1T = (__bf16*)alloc((size_t)Cc * 4 * Cc * 2);
  __bf16* il2W2T = (__bf16*)alloc((size_t)Cc * 4 * Cc * 2);
  __bf16* ol2W1T = (__bf16*)alloc((size_t)Cc * 4 * Cc * 2);
  __bf16* ol2W2T = (__bf16*)alloc((size_t)Cc * 4 * Cc * 2);

  const dim3 blk(256);
  const dim3 egrid(M * Cc / 256);

  auto wtrans = [&](const float* W, __bf16* WT, int Kd, int Nd) {
    int n = Kd * Nd;
    wtrans_kernel<<<dim3((n + 255) / 256), blk, 0, stream>>>(W, WT, Kd, Nd);
  };
  auto gemm = [&](int mode, const __bf16* Ain, const __bf16* WT, const float* bi,
                  const float* res, float* O, __bf16* Obf, int Kd, int Nd) {
    dim3 g(Nd / 256, M / 32);
    if (mode == 0)
      gemm_bf16_kernel<0><<<g, blk, 0, stream>>>(Ain, WT, bi, nullptr, O, Obf, Kd, Nd);
    else if (mode == 1)
      gemm_bf16_kernel<1><<<g, blk, 0, stream>>>(Ain, WT, bi, res, O, Obf, Kd, Nd);
    else
      gemm_bf16_kernel<2><<<g, blk, 0, stream>>>(Ain, WT, bi, nullptr, O, Obf, Kd, Nd);
  };
  auto two_layer = [&](const __bf16* inbf, const float* resf, const float** p,
                       const __bf16* W1T, const __bf16* W2T, float* O, int hidden) {
    gemm(0, inbf, W1T, p[1], nullptr, Hb, nullptr, Cc, hidden);
    ln_leaky_kernel<<<dim3(M), blk, 0, stream>>>(Hb, p[2], p[3], Hbf, hidden);
    gemm(1, Hbf, W2T, p[5], resf, O, nullptr, hidden, Cc);
  };

  // ---- prep: bf16 conversions / weight transposes ----
  f2bf_kernel<<<egrid, blk, 0, stream>>>(x, xbf);
  wtrans(il1[0], il1W1T, Cc, Cc);      wtrans(il1[4], il1W2T, Cc, Cc);
  wtrans(ol1[0], ol1W1T, Cc, Cc);      wtrans(ol1[4], ol1W2T, Cc, Cc);
  wtrans(fc_W, fcWT, 2 * Cc, Cc);
  wtrans(il2[0], il2W1T, Cc, 4 * Cc);  wtrans(il2[4], il2W2T, 4 * Cc, Cc);
  wtrans(ol2[0], ol2W1T, Cc, 4 * Cc);  wtrans(ol2[4], ol2W2T, 4 * Cc, Cc);

  // x1 = two_layer(x, il1); x1 *= mask; x1bf = bf16(x1)
  two_layer(xbf, x, il1, il1W1T, il1W2T, x1, Cc);
  mask_convert_kernel<<<egrid, blk, 0, stream>>>(x1, mask, x1bf, Cc);

  // sim -> top-9 graph -> max-aggregate -> interleaved features yin[M,512]
  sim_topk_agg_kernel<<<dim3(N / 16, B), blk, 0, stream>>>(x1bf, x1, mask, yin, yinbf);

  // yl = leaky(yin @ fc_W + fc_b)
  gemm(2, yinbf, fcWT, fc_b, nullptr, yl, ylbf, 2 * Cc, Cc);

  // tb = two_layer(yl, ol1); x2 = tb + x
  two_layer(ylbf, yl, ol1, ol1W1T, ol1W2T, tb, Cc);
  add_kernel<<<egrid, blk, 0, stream>>>(tb, x, x2, x2bf);

  // hb = two_layer(x2, il2) [hidden=4C]; hl = leaky(hb)
  two_layer(x2bf, x2, il2, il2W1T, il2W2T, hb, 4 * Cc);
  leaky_kernel<<<egrid, blk, 0, stream>>>(hb, hl, hlbf);

  // h2 = two_layer(hl, ol2) [hidden=4C]
  two_layer(hlbf, hl, ol2, ol2W1T, ol2W2T, h2, 4 * Cc);

  // out = (h2 + x2) * mask
  final_kernel<<<egrid, blk, 0, stream>>>(h2, x2, mask, outp, Cc);
}
